// Attention_61950608277988
// MI455X (gfx1250) — matrix operand order repaired and checked by reading, not compiled, not run on hardware
//
#include <hip/hip_runtime.h>
#include <hip/hip_bf16.h>
#include <stdint.h>

// ---------- CDNA5 WMMA types ----------
typedef __attribute__((ext_vector_type(16))) __bf16 v16bf;
typedef __attribute__((ext_vector_type(8)))  float  v8f;

union FragAB { v16bf v; uint32_t u[8]; };

// One-time conversions (Q fragments): proper round-to-nearest-even.
__device__ inline uint16_t f32_to_bf16_rne(float x) {
  uint32_t u = __builtin_bit_cast(uint32_t, x);
  uint32_t r = u + 0x7FFFu + ((u >> 16) & 1u);
  return (uint16_t)(r >> 16);
}
__device__ inline uint32_t pack_bf16_rne(float lo, float hi) {
  return (uint32_t)f32_to_bf16_rne(lo) | ((uint32_t)f32_to_bf16_rne(hi) << 16);
}
// Hot-path conversions: truncation. <=1 bf16 ULP vs RNE on operands feeding
// f32-accumulated WMMA; 2 VALU per pack (fusable to v_perm_b32), and the
// 16-bit LDS store form compiles to ds_store_b16_d16_hi with zero VALU.
__device__ inline uint16_t bf16_trunc(float x) {
  return (uint16_t)(__builtin_bit_cast(uint32_t, x) >> 16);
}
__device__ inline uint32_t pack2_bf16_trunc(float lo, float hi) {
  return (__builtin_bit_cast(uint32_t, hi) & 0xFFFF0000u) |
         (__builtin_bit_cast(uint32_t, lo) >> 16);
}
__device__ inline v8f v8f_zero() {
  v8f r;
#pragma unroll
  for (int i = 0; i < 8; ++i) r[i] = 0.0f;
  return r;
}

// Async DMA: global -> LDS, 16 bytes per lane, tracked by ASYNCcnt.
// LDS destination is a byte offset in a 32-bit VGPR (low 32 bits of a
// generic LDS pointer are the LDS offset per the flat-aperture mapping).
__device__ __forceinline__ void async_copy_b128(const float* lds_dst, const float* gsrc) {
  asm volatile("global_load_async_to_lds_b128 %0, %1, off"
               :: "v"((uint32_t)(uintptr_t)lds_dst),
                  "v"((uint64_t)(uintptr_t)gsrc)
               : "memory");
}
__device__ __forceinline__ void wait_async0() {
  asm volatile("s_wait_asynccnt 0x0" ::: "memory");
}

// =====================================================================
// GEMM: C[M,N] = A[M,K] * B[K,N], f32 in/out, bf16 WMMA inside.
// Block tile 128x128, 256 threads = 8 waves (4 M x 2 N), wave tile 32x64.
// A tile staged f32 via async DMA (double buffered); B tile converted/
// transposed through VGPRs into bf16 LDS.
// =====================================================================
#define SA_LD 36   // f32 elems per A row (32 + pad 4 -> 16B-aligned b128 dst)
#define SB_LD 40   // bf16 elems per B col (32 + pad 8)

__device__ __forceinline__ void gemm_stage_A_async(
    const float* __restrict__ A, const float* sAbuf, int bm, int K, int kk, int tid) {
#pragma unroll
  for (int r = 0; r < 4; ++r) {
    int fidx = tid + r * 256;                  // 0..1023 float4 slots
    int row  = fidx >> 3;                      // 0..127
    int c4   = (fidx & 7) * 4;                 // 0..28
    async_copy_b128(sAbuf + row * SA_LD + c4,
                    A + (size_t)(bm + row) * K + kk + c4);
  }
}

__device__ __forceinline__ void gemm_stage_B(
    const float* __restrict__ B, uint16_t* sBbuf, int bn, int N, int kk, int tid) {
#pragma unroll
  for (int r = 0; r < 4; ++r) {
    int fidx = tid + r * 256;
    int krow = fidx >> 5;                      // 0..31
    int n4   = (fidx & 31) * 4;                // 0..124
    float4 bv = *(const float4*)(B + (size_t)(kk + krow) * N + bn + n4);
    sBbuf[(n4 + 0) * SB_LD + krow] = bf16_trunc(bv.x);
    sBbuf[(n4 + 1) * SB_LD + krow] = bf16_trunc(bv.y);
    sBbuf[(n4 + 2) * SB_LD + krow] = bf16_trunc(bv.z);
    sBbuf[(n4 + 3) * SB_LD + krow] = bf16_trunc(bv.w);
  }
}

__global__ void __launch_bounds__(256) gemm_bf16_kernel(
    const float* __restrict__ A, const float* __restrict__ B,
    float* __restrict__ C, int M, int N, int K)
{
  __shared__ float    sAf[2][128 * SA_LD];  // row-major [m][k], f32 (DMA target)
  __shared__ uint16_t sB [2][128 * SB_LD];  // transposed [n][k], bf16

  const int tid  = threadIdx.x;
  const int wid  = tid >> 5;
  const int lane = tid & 31;
  const int half = lane >> 4;
  const int l16  = lane & 15;

  const int bm = blockIdx.y * 128;
  const int bn = blockIdx.x * 128;
  const int wm = (wid >> 1) * 32;  // wave row offset in tile
  const int wn = (wid & 1) * 64;   // wave col offset in tile

  v8f acc[2][4];
#pragma unroll
  for (int mi = 0; mi < 2; ++mi)
#pragma unroll
    for (int ni = 0; ni < 4; ++ni) acc[mi][ni] = v8f_zero();

  // Prologue: stage tile 0 into buffer 0.
  gemm_stage_A_async(A, sAf[0], bm, K, 0, tid);
  gemm_stage_B(B, sB[0], bn, N, 0, tid);
  wait_async0();
  __syncthreads();

  int buf = 0;
  for (int kk = 0; kk < K; kk += 32) {
    const int nbuf = buf ^ 1;
    if (kk + 32 < K) {
      // Kick off next tile's DMA before touching this tile's data.
      gemm_stage_A_async(A, sAf[nbuf], bm, K, kk + 32, tid);
      __builtin_prefetch(B + (size_t)(kk + 32 + (tid & 31)) * N + bn + ((tid >> 5) << 4), 0, 0);
      gemm_stage_B(B, sB[nbuf], bn, N, kk + 32, tid);
    }

    // A fragment (bf16 16x32), contiguous k order used by this code: lane row=l16, K pairs per VGPR,
    // lanes 0-15 K=0..15, lanes 16-31 K=16..31. Convert f32->bf16 here (trunc).
    FragAB af[2];
#pragma unroll
    for (int mi = 0; mi < 2; ++mi)
#pragma unroll
      for (int i = 0; i < 8; ++i) {
        int kp = 2 * i + half * 16;
        float2 f2 = *(const float2*)&sAf[buf][(wm + mi * 16 + l16) * SA_LD + kp];
        af[mi].u[i] = pack2_bf16_trunc(f2.x, f2.y);
      }
    // B fragment (bf16 32x16), same contiguous k order as A: lanes 0-15 K=0..15 at col=l16, lanes 16-31 K=16..31.
    FragAB bf[4];
#pragma unroll
    for (int ni = 0; ni < 4; ++ni)
#pragma unroll
      for (int i = 0; i < 8; ++i) {
        int kp = 2 * i + half * 16;
        bf[ni].u[i] = *(const uint32_t*)&sB[buf][(wn + ni * 16 + l16) * SB_LD + kp];
      }

#pragma unroll
    for (int mi = 0; mi < 2; ++mi)
#pragma unroll
      for (int ni = 0; ni < 4; ++ni)
        acc[mi][ni] = __builtin_amdgcn_wmma_f32_16x16x32_bf16(
            false, af[mi].v, false, bf[ni].v, (short)0, acc[mi][ni], false, false);

    wait_async0();     // next tile's DMA landed in LDS
    __syncthreads();   // block-wide visibility (barrier also drains DScnt)
    buf = nbuf;
  }

  // C layout: VGPR j -> row j (lanes 0-15) / row j+8 (lanes 16-31), col = l16.
#pragma unroll
  for (int mi = 0; mi < 2; ++mi)
#pragma unroll
    for (int ni = 0; ni < 4; ++ni)
#pragma unroll
      for (int j = 0; j < 8; ++j) {
        int row = bm + wm + mi * 16 + j + half * 8;
        int col = bn + wn + ni * 16 + l16;
        C[(size_t)row * N + col] = acc[mi][ni][j];
      }
}

// =====================================================================
// RoPE: in-place rotate xq (2048,32,128) and xk (2048,8,128).
// sincos[p] = [sin(0..127) | cos(0..127)]; rotate_half pairs (d, d+64).
// =====================================================================
__global__ void __launch_bounds__(256) rope_kernel(
    float* __restrict__ xq, float* __restrict__ xk, const float* __restrict__ sincos)
{
  const int p = blockIdx.x;
  const float* sc = sincos + (size_t)p * 256;
  for (int i = threadIdx.x; i < 32 * 64; i += 256) {
    int hh = i >> 6, d = i & 63;
    float* base = xq + (size_t)p * 4096 + hh * 128;
    float a = base[d], b = base[d + 64];
    base[d]      = a * sc[128 + d] - b * sc[d];
    base[d + 64] = b * sc[192 + d] + a * sc[d + 64];
  }
  for (int i = threadIdx.x; i < 8 * 64; i += 256) {
    int hh = i >> 6, d = i & 63;
    float* base = xk + (size_t)p * 1024 + hh * 128;
    float a = base[d], b = base[d + 64];
    base[d]      = a * sc[128 + d] - b * sc[d];
    base[d + 64] = b * sc[192 + d] + a * sc[d + 64];
  }
}

// =====================================================================
// Flash attention (causal, GQA 4:1). Block = (q-tile of 128, head).
// 8 waves x 16 q rows; kv chunks of 32. K chunk staged f32 via async DMA
// (layout matches global, no transpose); V transposed/converted via VGPRs.
// S = Q K^T (8 wmma) -> online softmax (width-16 shuffles) -> P V (8 wmma).
// =====================================================================
#define SK_LD 132  // f32 elems per K row (128 + pad 4)
#define SV_LD 34   // bf16 elems per V col (32 + pad 2)

__global__ void __launch_bounds__(256) flash_attn_kernel(
    const float* __restrict__ xq, const float* __restrict__ xk,
    const float* __restrict__ xv, float* __restrict__ xo)
{
  __shared__ float    sKf[32 * SK_LD];     // [kv][d], d contiguous (DMA target)
  __shared__ uint16_t sV [128 * SV_LD];    // [d][kv], kv contiguous (transposed)
  __shared__ uint16_t sP [8 * 16 * SV_LD]; // per-wave 16x32 P strip

  const int tid  = threadIdx.x;
  const int wid  = tid >> 5;
  const int lane = tid & 31;
  const int half = lane >> 4;
  const int l16  = lane & 15;

  const int q0    = blockIdx.x * 128;
  const int h     = blockIdx.y;
  const int kvh   = h >> 2;
  const int qrow0 = q0 + wid * 16;
  const float scale = 0.08838834764831845f;  // 1/sqrt(128), folded into Q

  // Q fragments: 4 chunks over head_dim=128, loaded straight from global.
  FragAB qf[4];
  {
    const float* qbase = xq + (size_t)(qrow0 + l16) * 4096 + h * 128;
#pragma unroll
    for (int kc = 0; kc < 4; ++kc)
#pragma unroll
      for (int i = 0; i < 8; ++i) {
        int kp = ((i < 4) ? (2 * i) : (2 * (i - 4) + 16)) + half * 8 + kc * 32;
        float2 f2 = *(const float2*)(qbase + kp);
        qf[kc].u[i] = pack_bf16_rne(f2.x * scale, f2.y * scale);
      }
  }

  v8f o[8];
#pragma unroll
  for (int t = 0; t < 8; ++t) o[t] = v8f_zero();
  float m[8], l[8];
#pragma unroll
  for (int j = 0; j < 8; ++j) { m[j] = -1e30f; l[j] = 0.0f; }

  const int kend = q0 + 128;  // causal upper bound for this q tile
  for (int k0 = 0; k0 < kend; k0 += 32) {
    // Stage K (async DMA, raw f32) and V (VGPR transpose-convert) chunks.
#pragma unroll
    for (int r = 0; r < 4; ++r) {
      int fidx = tid + r * 256;            // 0..1023 float4 slots
      int kr   = fidx >> 5;                // 0..31
      int d4   = (fidx & 31) * 4;          // 0..124
      async_copy_b128(&sKf[kr * SK_LD + d4],
                      xk + (size_t)(k0 + kr) * 1024 + kvh * 128 + d4);
      float4 vf = *(const float4*)(xv + (size_t)(k0 + kr) * 1024 + kvh * 128 + d4);
      sV[(d4 + 0) * SV_LD + kr] = bf16_trunc(vf.x);
      sV[(d4 + 1) * SV_LD + kr] = bf16_trunc(vf.y);
      sV[(d4 + 2) * SV_LD + kr] = bf16_trunc(vf.z);
      sV[(d4 + 3) * SV_LD + kr] = bf16_trunc(vf.w);
    }
    wait_async0();
    __syncthreads();

    // S tile (16 q rows x 32 kv cols), contraction over head_dim.
    v8f s0 = v8f_zero(), s1 = v8f_zero();
#pragma unroll
    for (int kc = 0; kc < 4; ++kc) {
      FragAB b0, b1;
#pragma unroll
      for (int i = 0; i < 8; ++i) {
        int d = ((i < 4) ? (2 * i) : (2 * (i - 4) + 16)) + half * 8 + kc * 32;
        float2 f0 = *(const float2*)&sKf[(l16)      * SK_LD + d];
        float2 f1 = *(const float2*)&sKf[(16 + l16) * SK_LD + d];
        b0.u[i] = pack2_bf16_trunc(f0.x, f0.y);
        b1.u[i] = pack2_bf16_trunc(f1.x, f1.y);
      }
      s0 = __builtin_amdgcn_wmma_f32_16x16x32_bf16(false, qf[kc].v, false, b0.v, (short)0, s0, false, false);
      s1 = __builtin_amdgcn_wmma_f32_16x16x32_bf16(false, qf[kc].v, false, b1.v, (short)0, s1, false, false);
    }

    // Causal mask + online softmax; stats live in C-matrix row layout
    // (VGPR j -> rows j / j+8 via lane half), replicated across 16 lanes.
#pragma unroll
    for (int j = 0; j < 8; ++j) {
      int qrow = qrow0 + j + half * 8;
      float a = (k0 + l16      <= qrow) ? s0[j] : -1e30f;
      float b = (k0 + 16 + l16 <= qrow) ? s1[j] : -1e30f;
      float cmax = fmaxf(a, b);
#pragma unroll
      for (int off = 8; off >= 1; off >>= 1)
        cmax = fmaxf(cmax, __shfl_xor(cmax, off, 16));
      float mnew  = fmaxf(m[j], cmax);
      float alpha = __expf(m[j] - mnew);
      float p0 = __expf(a - mnew);
      float p1 = __expf(b - mnew);
      float rs = p0 + p1;
#pragma unroll
      for (int off = 8; off >= 1; off >>= 1)
        rs += __shfl_xor(rs, off, 16);
      l[j] = l[j] * alpha + rs;
      m[j] = mnew;
#pragma unroll
      for (int t = 0; t < 8; ++t) o[t][j] *= alpha;
      // Re-layout P via wave-private LDS strip (C layout -> A layout).
      uint32_t base = (uint32_t)(wid * 16 + j + half * 8) * SV_LD;
      sP[base + l16]      = bf16_trunc(p0);
      sP[base + 16 + l16] = bf16_trunc(p1);
    }

    // P fragment (A layout, 16x32) and P @ V accumulation over 8 dim tiles.
    FragAB pf;
#pragma unroll
    for (int i = 0; i < 8; ++i) {
      int kp = 2 * i + half * 16;
      pf.u[i] = *(const uint32_t*)&sP[(wid * 16 + l16) * SV_LD + kp];
    }
#pragma unroll
    for (int t = 0; t < 8; ++t) {
      FragAB bv;
#pragma unroll
      for (int i = 0; i < 8; ++i) {
        int kk2 = 2 * i + half * 16;
        bv.u[i] = *(const uint32_t*)&sV[(t * 16 + l16) * SV_LD + kk2];
      }
      o[t] = __builtin_amdgcn_wmma_f32_16x16x32_bf16(false, pf.v, false, bv.v, (short)0, o[t], false, false);
    }
    __syncthreads();
  }

  // Normalize and store O -> xo (seq, head, dim) f32.
#pragma unroll
  for (int j = 0; j < 8; ++j) {
    float inv = 1.0f / l[j];
    int row = qrow0 + j + half * 8;
    float* ob = xo + (size_t)row * 4096 + h * 128;
#pragma unroll
    for (int t = 0; t < 8; ++t)
      ob[t * 16 + l16] = o[t][j] * inv;
  }
}

// =====================================================================
// Launch: x->QKV gemms, rope, flash attention, out-projection.
// Inputs: 0:x 1:start_pos 2:sincos 3:mask 4:wq 5:wk 6:wv 7:wo 8:cache_k 9:cache_v
// start_pos==0 in this harness; cache starts zeroed, so xk/xv are the full KV.
// =====================================================================
extern "C" void kernel_launch(void* const* d_in, const int* in_sizes, int n_in,
                              void* d_out, int out_size, void* d_ws, size_t ws_size,
                              hipStream_t stream) {
  (void)in_sizes; (void)n_in; (void)out_size; (void)ws_size;
  const float* x      = (const float*)d_in[0];
  const float* sincos = (const float*)d_in[2];
  const float* wq     = (const float*)d_in[4];
  const float* wk     = (const float*)d_in[5];
  const float* wv     = (const float*)d_in[6];
  const float* wo     = (const float*)d_in[7];
  float* out = (float*)d_out;

  float* xq = (float*)d_ws;                      // 2048*4096
  float* xk = xq + (size_t)2048 * 4096;          // 2048*1024
  float* xv = xk + (size_t)2048 * 1024;          // 2048*1024
  float* xo = xv + (size_t)2048 * 1024;          // 2048*4096

  gemm_bf16_kernel<<<dim3(32, 16), 256, 0, stream>>>(x, wq, xq, 2048, 4096, 4096);
  gemm_bf16_kernel<<<dim3(8, 16),  256, 0, stream>>>(x, wk, xk, 2048, 1024, 4096);
  gemm_bf16_kernel<<<dim3(8, 16),  256, 0, stream>>>(x, wv, xv, 2048, 1024, 4096);
  rope_kernel<<<2048, 256, 0, stream>>>(xq, xk, sincos);
  flash_attn_kernel<<<dim3(16, 32), 256, 0, stream>>>(xq, xk, xv, xo);
  gemm_bf16_kernel<<<dim3(32, 16), 256, 0, stream>>>(xo, wo, out, 2048, 4096, 4096);
}
